// MeshGraFormer_2465311228645
// MI455X (gfx1250) — matrix operand order repaired and checked by reading, not compiled, not run on hardware
//
#include <hip/hip_runtime.h>
#include <hip/hip_bf16.h>
#include <math.h>

// ---------------------------------------------------------------------------
// Types for CDNA5 WMMA
// ---------------------------------------------------------------------------
typedef _Float16 v16h __attribute__((ext_vector_type(16)));
typedef _Float16 h2   __attribute__((ext_vector_type(2)));
typedef float    v8f  __attribute__((ext_vector_type(8)));

// ---------------------------------------------------------------------------
// Flexible batched GEMM:  C = act( alpha * A*B + biasCol + biasRow [+ C] )
// A: logical M x K row-major (no transpose; weights are pre-transposed on host)
// B: logical K x N; TB=1 reads B[n*ldb + k] (transposed operand, attention k^T)
// Batch z offsets: ptr + (z/cnt)*outer + (z%cnt)*inner  (per A, B, C)
// fp32 in memory, f16 tiles in LDS, f32 accumulation via v_wmma_f32_16x16x32_f16
// Fast paths: full tiles + aligned ld -> float4 global loads, h2 LDS stores.
// Slow path: branchless min-clamped loads (no exec-mask serialization).
// ---------------------------------------------------------------------------
struct GemmP {
  const float* A; const float* B; float* C;
  const float* biasCol; const float* biasRow;
  long long aOuter, aInner, bOuter, bInner, cOuter, cInner;
  int aCnt, bCnt, cCnt;
  int lda, ldb, ldc;
  int M, N, K;
  float alpha;
  int accum, relu;
};

#define BM 64
#define BN 64
#define BK 32
#define LDS_STRIDE 34  // pad: keeps half2 pairs 4B-aligned, staggers banks

template <int TB>
__global__ __launch_bounds__(128) void gemm_wmma(GemmP p) {
  __shared__ _Float16 sA[BM * LDS_STRIDE];
  __shared__ _Float16 sB[BN * LDS_STRIDE];

  const int z = blockIdx.z;
  const float* Ab = p.A + (long long)(z / p.aCnt) * p.aOuter + (long long)(z % p.aCnt) * p.aInner;
  const float* Bb = p.B + (long long)(z / p.bCnt) * p.bOuter + (long long)(z % p.bCnt) * p.bInner;
  float*       Cb = p.C + (long long)(z / p.cCnt) * p.cOuter + (long long)(z % p.cCnt) * p.cInner;

  const int m0 = blockIdx.y * BM;
  const int n0 = blockIdx.x * BN;
  const int tid  = threadIdx.x;
  const int lane = tid & 31;
  const int w    = tid >> 5;
  const int wm   = (w >> 1) * 32;  // wave's M offset in block tile
  const int wn   = (w & 1) * 32;   // wave's N offset in block tile
  const int mlane = lane & 15;
  const int hi    = lane >> 4;     // 0: lanes 0-15, 1: lanes 16-31

  const bool aFast = ((p.lda & 3) == 0) && (m0 + BM <= p.M);
  const bool bFast = ((p.ldb & 3) == 0) && (n0 + BN <= p.N);
  const int ldav = p.lda >> 2;
  const int ldbv = p.ldb >> 2;
  const int kFull = p.K / BK;      // number of complete K steps

  v8f acc[2][2] = {};

  const int kSteps = (p.K + BK - 1) / BK;
  for (int ks = 0; ks < kSteps; ++ks) {
    const int k0 = ks * BK;
    const bool fullK = ks < kFull;

    // ---------------- stage A tile (BM x BK) as f16, row-major ----------------
    if (aFast && fullK) {
      const float4* A4 = (const float4*)(Ab + (long long)m0 * p.lda + (long long)k0);
      #pragma unroll
      for (int r = 0; r < 4; ++r) {
        int f = tid + r * 128;            // 0..511 float4s
        int row = f >> 3, c4 = f & 7;     // 8 float4 per 32-wide row
        float4 v = A4[(long long)row * ldav + c4];
        int base = row * LDS_STRIDE + c4 * 4;
        h2 p01, p23;
        p01.x = (_Float16)v.x; p01.y = (_Float16)v.y;
        p23.x = (_Float16)v.z; p23.y = (_Float16)v.w;
        *(h2*)&sA[base]     = p01;
        *(h2*)&sA[base + 2] = p23;
      }
    } else {
      #pragma unroll
      for (int r = 0; r < 16; ++r) {
        int lin = tid + r * 128;
        int mm = lin >> 5, kk = lin & 31;
        int gm = m0 + mm, gk = k0 + kk;
        bool ok = (gm < p.M) && (gk < p.K);
        int gmc = min(gm, p.M - 1);
        int gkc = min(gk, p.K - 1);
        float v = Ab[(long long)gmc * p.lda + gkc];   // unconditional, clamped
        v = ok ? v : 0.f;
        sA[mm * LDS_STRIDE + kk] = (_Float16)v;
      }
    }

    // ---------------- stage B tile (BK x BN) as f16, stored n-major ----------------
    if (bFast && fullK) {
      if (TB) {
        const float4* B4 = (const float4*)(Bb + (long long)n0 * p.ldb + (long long)k0);
        #pragma unroll
        for (int r = 0; r < 4; ++r) {
          int f = tid + r * 128;
          int n = f >> 3, k4 = f & 7;
          float4 v = B4[(long long)n * ldbv + k4];
          int base = n * LDS_STRIDE + k4 * 4;
          h2 p01, p23;
          p01.x = (_Float16)v.x; p01.y = (_Float16)v.y;
          p23.x = (_Float16)v.z; p23.y = (_Float16)v.w;
          *(h2*)&sB[base]     = p01;
          *(h2*)&sB[base + 2] = p23;
        }
      } else {
        const float4* B4 = (const float4*)(Bb + (long long)k0 * p.ldb + (long long)n0);
        #pragma unroll
        for (int r = 0; r < 4; ++r) {
          int f = tid + r * 128;
          int k = f >> 4, n4 = f & 15;    // 16 float4 per 64-wide row
          float4 v = B4[(long long)k * ldbv + n4];
          int nb = n4 * 4;
          sB[(nb + 0) * LDS_STRIDE + k] = (_Float16)v.x;
          sB[(nb + 1) * LDS_STRIDE + k] = (_Float16)v.y;
          sB[(nb + 2) * LDS_STRIDE + k] = (_Float16)v.z;
          sB[(nb + 3) * LDS_STRIDE + k] = (_Float16)v.w;
        }
      }
    } else {
      #pragma unroll
      for (int r = 0; r < 16; ++r) {
        int lin = tid + r * 128;
        int nn = lin >> 5, kk = lin & 31;
        int gn = n0 + nn, gk = k0 + kk;
        bool ok = (gn < p.N) && (gk < p.K);
        int gnc = min(gn, p.N - 1);
        int gkc = min(gk, p.K - 1);
        float v = TB ? Bb[(long long)gnc * p.ldb + gkc]
                     : Bb[(long long)gkc * p.ldb + gnc];
        v = ok ? v : 0.f;
        sB[nn * LDS_STRIDE + kk] = (_Float16)v;
      }
    }

    // prefetch next K tiles into cache (global_prefetch_b8)
    if (ks + 1 < kSteps) {
      int nk = k0 + BK;
      __builtin_prefetch((const void*)(Ab + (long long)m0 * p.lda + nk), 0, 1);
      __builtin_prefetch(TB ? (const void*)(Bb + (long long)n0 * p.ldb + nk)
                            : (const void*)(Bb + (long long)nk * p.ldb + n0), 0, 1);
    }
    __syncthreads();

    // ---- build fragments per ISA 7.12.2 layouts ----
    v16h af[2], bf[2];
    #pragma unroll
    for (int t = 0; t < 2; ++t) {
      const int mrow = wm + t * 16 + mlane;
      #pragma unroll
      for (int pj = 0; pj < 8; ++pj) {
        // A 16x32 f16: lanes<16: K pairs 0,2,4,6,16,18,20,22 ; lanes>=16: +8
        int kk = 2 * pj + (pj < 4 ? 0 : 8) + (hi ? 8 : 0);
        h2 hv = *(const h2*)&sA[mrow * LDS_STRIDE + kk];
        af[t][2 * pj]     = hv.x;
        af[t][2 * pj + 1] = hv.y;
      }
      const int ncol = wn + t * 16 + mlane;
      #pragma unroll
      for (int pj = 0; pj < 8; ++pj) {
        // B 32x16 f16: lanes<16: K 0..7,16..23 ; lanes>=16: +8 (matches A)
        int kk = 2 * pj + (pj < 4 ? 0 : 8) + (hi ? 8 : 0);
        h2 hv = *(const h2*)&sB[ncol * LDS_STRIDE + kk];
        bf[t][2 * pj]     = hv.x;
        bf[t][2 * pj + 1] = hv.y;
      }
    }

    #pragma unroll
    for (int mt = 0; mt < 2; ++mt)
      #pragma unroll
      for (int nt = 0; nt < 2; ++nt)
        acc[mt][nt] = __builtin_amdgcn_wmma_f32_16x16x32_f16(
            false, af[mt], false, bf[nt], (short)0, acc[mt][nt], false, false);

    __syncthreads();
  }

  // ---- epilogue: scale, bias, accumulate, relu, store ----
  if ((m0 + BM <= p.M) && (n0 + BN <= p.N)) {
    #pragma unroll
    for (int mt = 0; mt < 2; ++mt) {
      #pragma unroll
      for (int nt = 0; nt < 2; ++nt) {
        const int gn = n0 + wn + nt * 16 + mlane;
        float bc = p.biasCol ? p.biasCol[gn] : 0.f;
        #pragma unroll
        for (int r = 0; r < 8; ++r) {
          const int gm = m0 + wm + mt * 16 + r + (hi << 3);
          float v = acc[mt][nt][r] * p.alpha + bc;
          if (p.biasRow) v += p.biasRow[gm];
          long long ci = (long long)gm * p.ldc + gn;
          if (p.accum) v += Cb[ci];
          if (p.relu) v = fmaxf(v, 0.f);
          Cb[ci] = v;
        }
      }
    }
  } else {
    #pragma unroll
    for (int mt = 0; mt < 2; ++mt) {
      #pragma unroll
      for (int nt = 0; nt < 2; ++nt) {
        const int gn = n0 + wn + nt * 16 + mlane;
        #pragma unroll
        for (int r = 0; r < 8; ++r) {
          const int gm = m0 + wm + mt * 16 + r + (hi << 3);
          if (gm < p.M && gn < p.N) {
            float v = acc[mt][nt][r] * p.alpha;
            if (p.biasCol) v += p.biasCol[gn];
            if (p.biasRow) v += p.biasRow[gm];
            long long ci = (long long)gm * p.ldc + gn;
            if (p.accum) v += Cb[ci];
            if (p.relu) v = fmaxf(v, 0.f);
            Cb[ci] = v;
          }
        }
      }
    }
  }
}

// ---------------------------------------------------------------------------
// LayerNorm over last dim (=128) with std ddof=1 + eps (GraFormer style)
// ---------------------------------------------------------------------------
__global__ __launch_bounds__(128) void layernorm128(const float* x, const float* a,
                                                    const float* b, float* y, int rows) {
  __shared__ float red[128];
  const long long row = blockIdx.x;
  const int t = threadIdx.x;
  float v = x[row * 128 + t];
  red[t] = v; __syncthreads();
  for (int s = 64; s > 0; s >>= 1) { if (t < s) red[t] += red[t + s]; __syncthreads(); }
  float mean = red[0] * (1.f / 128.f); __syncthreads();
  float d = v - mean;
  red[t] = d * d; __syncthreads();
  for (int s = 64; s > 0; s >>= 1) { if (t < s) red[t] += red[t + s]; __syncthreads(); }
  float stdv = sqrtf(red[0] * (1.f / 127.f));
  y[row * 128 + t] = a[t] * d / (stdv + 1e-6f) + b[t];
  (void)rows;
}

// ---------------------------------------------------------------------------
// Row softmax (row width n <= 256)
// ---------------------------------------------------------------------------
__global__ __launch_bounds__(256) void softmax_rows(float* s, int n) {
  __shared__ float red[256];
  const long long row = blockIdx.x;
  const int t = threadIdx.x;
  float v = (t < n) ? s[row * n + t] : -1e30f;
  red[t] = v; __syncthreads();
  for (int st = 128; st > 0; st >>= 1) { if (t < st) red[t] = fmaxf(red[t], red[t + st]); __syncthreads(); }
  float mx = red[0]; __syncthreads();
  float e = (t < n) ? __expf(v - mx) : 0.f;
  red[t] = e; __syncthreads();
  for (int st = 128; st > 0; st >>= 1) { if (t < st) red[t] += red[t + st]; __syncthreads(); }
  if (t < n) s[row * n + t] = e / red[0];
}

// ---------------------------------------------------------------------------
// Cheb:  d[i] = rsqrt(sum_j adj[i][j]) ;  L = I - d_i * adj * d_j   (stride ldp)
// ---------------------------------------------------------------------------
__global__ __launch_bounds__(256) void row_rsqrt_deg(const float* adj, float* d, int n) {
  __shared__ float red[256];
  const int i = blockIdx.x, t = threadIdx.x;
  float s = 0.f;
  for (int j = t; j < n; j += 256) s += adj[(long long)i * n + j];
  red[t] = s; __syncthreads();
  for (int st = 128; st > 0; st >>= 1) { if (t < st) red[t] += red[t + st]; __syncthreads(); }
  if (t == 0) d[i] = rsqrtf(red[0]);
}

__global__ __launch_bounds__(256) void build_cheb_L(const float* adj, const float* d,
                                                    float* L, int n, int ldp) {
  long long idx = (long long)blockIdx.x * 256 + threadIdx.x;
  if (idx < (long long)n * ldp) {
    int i = (int)(idx / ldp), j = (int)(idx % ldp);
    float v = 0.f;
    if (j < n) {
      v = -d[i] * adj[(long long)i * n + j] * d[j];
      if (i == j) v += 1.f;
    }
    L[idx] = v;
  }
}

// GraphNet:  d[j] = rsqrt(colsum + 1e-5) ;  L = d_i * A * d_j   (stride ldp)
__global__ __launch_bounds__(256) void col_rsqrt_deg(const float* a, float* d, int n) {
  __shared__ float red[256];
  const int j = blockIdx.x, t = threadIdx.x;
  float s = 0.f;
  for (int i = t; i < n; i += 256) s += a[(long long)i * n + j];
  red[t] = s; __syncthreads();
  for (int st = 128; st > 0; st >>= 1) { if (t < st) red[t] += red[t + st]; __syncthreads(); }
  if (t == 0) d[j] = rsqrtf(red[0] + 1e-5f);
}

__global__ __launch_bounds__(256) void build_gnet_L(const float* a, const float* d,
                                                    float* L, int n, int ldp) {
  long long idx = (long long)blockIdx.x * 256 + threadIdx.x;
  if (idx < (long long)n * ldp) {
    int i = (int)(idx / ldp), j = (int)(idx % ldp);
    L[idx] = (j < n) ? d[i] * a[(long long)i * n + j] * d[j] : 0.f;
  }
}

// out[c*ldo + r] = in[r*cols_in + c]  (transpose with padded output stride)
__global__ __launch_bounds__(256) void transpose_pad(const float* in, float* out,
                                                     int rows_in, int cols_in, int ldo) {
  long long idx = (long long)blockIdx.x * 256 + threadIdx.x;
  if (idx < (long long)rows_in * cols_in) {
    int r = (int)(idx / cols_in), c = (int)(idx % cols_in);
    out[(long long)c * ldo + r] = in[idx];
  }
}

__global__ __launch_bounds__(256) void add_inplace(float* a, const float* b, long long cnt) {
  long long i = (long long)blockIdx.x * 256 + threadIdx.x;
  if (i < cnt) a[i] += b[i];
}

// ---------------------------------------------------------------------------
// Host orchestration
// ---------------------------------------------------------------------------
extern "C" void kernel_launch(void* const* d_in, const int* in_sizes, int n_in,
                              void* d_out, int out_size, void* d_ws, size_t ws_size,
                              hipStream_t stream) {
  (void)out_size; (void)ws_size;
  if (n_in < 90) return;

  const int LEV[4] = {21, 49, 194, 778};
  const int LDP[4] = {24, 52, 196, 780};   // padded row strides (mult of 4)
  const int B = 256, DK = 32, H = 4;

  auto F = [&](int i) -> const float* { return (const float*)d_in[i]; };

  // --- input index maps (JAX sorted-key flatten vs insertion order) ---
  struct LayerIdx {
    int A_hat, bk, bo, bq, bv, g1W, g1b, g2W, g2b, ln1a, ln1b, ln2a, ln2b,
        r1c1W, r1c1b, r1c2W, r1c2b, r2c1W, r2c1b, r2c2W, r2c2b, upW, upb,
        wk, wo, wq, wv;
  } LI[3];
  int adjI[4], ginW, ginB, goutW, goutB, xI;
  const bool sortedOrder = (in_sizes[0] == 441);  // adjs first (sorted keys) vs x first
  if (sortedOrder) {
    for (int k = 0; k < 4; ++k) adjI[k] = k;
    ginW = 4; ginB = 5; goutW = 6; goutB = 7; xI = 89;
    for (int l = 0; l < 3; ++l) {
      int b = 8 + 27 * l;
      LI[l] = { b+0,b+1,b+2,b+3,b+4,b+5,b+6,b+7,b+8,b+9,b+10,b+11,b+12,
                b+13,b+14,b+15,b+16,b+17,b+18,b+19,b+20,b+21,b+22,b+23,b+24,b+25,b+26 };
    }
  } else {
    xI = 0; for (int k = 0; k < 4; ++k) adjI[k] = 1 + k;
    ginW = 5; ginB = 6; goutW = 7; goutB = 8;
    for (int l = 0; l < 3; ++l) {
      int b = 9 + 27 * l;
      LI[l].ln1a=b+0;  LI[l].ln1b=b+1;  LI[l].wq=b+2;   LI[l].bq=b+3;
      LI[l].wk=b+4;    LI[l].bk=b+5;    LI[l].wv=b+6;   LI[l].bv=b+7;
      LI[l].wo=b+8;    LI[l].bo=b+9;    LI[l].ln2a=b+10; LI[l].ln2b=b+11;
      LI[l].A_hat=b+12; LI[l].g1W=b+13; LI[l].g1b=b+14; LI[l].g2W=b+15; LI[l].g2b=b+16;
      LI[l].r1c1W=b+17; LI[l].r1c1b=b+18; LI[l].r1c2W=b+19; LI[l].r1c2b=b+20;
      LI[l].r2c1W=b+21; LI[l].r2c1b=b+22; LI[l].r2c2W=b+23; LI[l].r2c2b=b+24;
      LI[l].upW=b+25;   LI[l].upb=b+26;
    }
  }

  // --- workspace layout (floats); all blocks stay 16B-aligned ---
  float* ws = (float*)d_ws;
  size_t o = 0;
  float* Lc[4];
  for (int i = 0; i < 4; ++i) { Lc[i] = ws + o; o += (size_t)LEV[i] * LDP[i]; }
  float* dvec = ws + o; o += 780;
  float* Lg   = ws + o; o += (size_t)194 * 196;
  float* upWT = ws + o; o += (size_t)778 * 196;   // transposed+padded unpool weight
  const size_t ACT = (size_t)B * 778 * 128;       // 25,493,504 floats
  float* act  = ws + o; o += ACT;
  float* pbuf = ws + o; o += ACT;
  float* b1   = ws + o; o += ACT;
  float* b2   = ws + o; o += ACT;
  float* h3   = ws + o; o += (size_t)B * 194 * 256 + 400000;  // pad so S fits
  float* Sb   = b2;  // scores alias b2+h3 region (both dead during attention)

  // --- GEMM launcher; transB selects template instantiation ---
  auto gemm = [&](int M, int N, int K,
                  const float* A, int lda, long long aO, long long aI, int aC,
                  const float* Bm, int ldb, int transB, long long bO, long long bI, int bC,
                  float* Cm, int ldc, long long cO, long long cI, int cC,
                  const float* bCol, const float* bRow,
                  float alpha, int accum, int relu, int batch) {
    GemmP p;
    p.A = A; p.B = Bm; p.C = Cm; p.biasCol = bCol; p.biasRow = bRow;
    p.aOuter = aO; p.aInner = aI; p.bOuter = bO; p.bInner = bI; p.cOuter = cO; p.cInner = cI;
    p.aCnt = aC; p.bCnt = bC; p.cCnt = cC;
    p.lda = lda; p.ldb = ldb; p.ldc = ldc;
    p.M = M; p.N = N; p.K = K; p.alpha = alpha;
    p.accum = accum; p.relu = relu;
    dim3 g((unsigned)((N + 63) / 64), (unsigned)((M + 63) / 64), (unsigned)batch);
    if (transB) hipLaunchKernelGGL(gemm_wmma<1>, g, dim3(128), 0, stream, p);
    else        hipLaunchKernelGGL(gemm_wmma<0>, g, dim3(128), 0, stream, p);
  };

  // --- precompute Cheb Laplacians for all 4 levels (padded strides) ---
  for (int i = 0; i < 4; ++i) {
    int n = LEV[i], ldp = LDP[i];
    row_rsqrt_deg<<<n, 256, 0, stream>>>(F(adjI[i]), dvec, n);
    long long tot = (long long)n * ldp;
    build_cheb_L<<<(unsigned)((tot + 255) / 256), 256, 0, stream>>>(F(adjI[i]), dvec, Lc[i], n, ldp);
  }

  float* curr = act;
  float* spare = pbuf;

  // --- input ChebConv: (B,21,2) -> (B,21,128) ---
  {
    const float* x = F(xI);
    gemm(21, 2, 21, Lc[0], LDP[0], 0, 0, 1,
         x, 2, 0, 21 * 2, 0, 1,
         b1, 2, 21 * 2, 0, 1, nullptr, nullptr, 1.f, 0, 0, B);            // t = L0@x
    gemm(B * 21, 128, 2, x, 2, 0, 0, 1,
         F(ginW), 128, 0, 0, 0, 1,
         curr, 128, 0, 0, 1, F(ginB), nullptr, 1.f, 0, 0, 1);             // curr = x@W0+b
    gemm(B * 21, 128, 2, b1, 2, 0, 0, 1,
         F(ginW) + 2 * 128, 128, 0, 0, 0, 1,
         curr, 128, 0, 0, 1, nullptr, nullptr, 1.f, 1, 0, 1);             // curr += t@W1
  }

  for (int l = 0; l < 3; ++l) {
    const int n = LEV[l];
    const int nnx = LEV[l + 1];
    const int rows = B * n;
    const long long s128 = (long long)n * 128;
    const LayerIdx& Ld = LI[l];
    const size_t qs = (size_t)B * n * 128;

    // ================= MHA (pre-LN, residual) =================
    layernorm128<<<(unsigned)rows, 128, 0, stream>>>(curr, F(Ld.ln1a), F(Ld.ln1b), b2, rows);
    float* Q = b1; float* Kb = b1 + qs; float* V = b1 + 2 * qs; float* O = b1 + 3 * qs;
    gemm(rows, 128, 128, b2, 128, 0, 0, 1, F(Ld.wq), 128, 0, 0, 0, 1,
         Q, 128, 0, 0, 1, F(Ld.bq), nullptr, 1.f, 0, 0, 1);
    gemm(rows, 128, 128, b2, 128, 0, 0, 1, F(Ld.wk), 128, 0, 0, 0, 1,
         Kb, 128, 0, 0, 1, F(Ld.bk), nullptr, 1.f, 0, 0, 1);
    gemm(rows, 128, 128, b2, 128, 0, 0, 1, F(Ld.wv), 128, 0, 0, 0, 1,
         V, 128, 0, 0, 1, F(Ld.bv), nullptr, 1.f, 0, 0, 1);
    // scores[b,h] = (q_bh @ k_bh^T) / sqrt(dk)   — batch z = b*H + h via strides
    gemm(n, n, DK,
         Q, 128, s128, DK, H,
         Kb, 128, 1, s128, DK, H,
         Sb, n, (long long)n * n, 0, 1,
         nullptr, nullptr, 0.17677669529663687f, 0, 0, B * H);
    softmax_rows<<<(unsigned)(B * H * n), 256, 0, stream>>>(Sb, n);
    // o[b,h] = attn @ v_bh  -> written straight back into (B,N,H,dk) layout
    gemm(n, DK, n,
         Sb, n, (long long)n * n, 0, 1,
         V, 128, 0, s128, DK, H,
         O, 128, s128, DK, H,
         nullptr, nullptr, 1.f, 0, 0, B * H);
    gemm(rows, 128, 128, O, 128, 0, 0, 1, F(Ld.wo), 128, 0, 0, 0, 1,
         curr, 128, 0, 0, 1, F(Ld.bo), nullptr, 1.f, 1, 0, 1);            // curr += O@wo+bo

    // ================= GraphNet (pre-LN, residual) =================
    layernorm128<<<(unsigned)rows, 128, 0, stream>>>(curr, F(Ld.ln2a), F(Ld.ln2b), b2, rows);
    col_rsqrt_deg<<<n, 256, 0, stream>>>(F(Ld.A_hat), dvec, n);
    {
      long long tot = (long long)n * 196;
      build_gnet_L<<<(unsigned)((tot + 255) / 256), 256, 0, stream>>>(F(Ld.A_hat), dvec, Lg, n, 196);
    }
    float* t1 = b2 + qs;
    gemm(n, 128, n, Lg, 196, 0, 0, 1, b2, 128, 0, s128, 0, 1,
         t1, 128, s128, 0, 1, nullptr, nullptr, 1.f, 0, 0, B);
    gemm(rows, 256, 128, t1, 128, 0, 0, 1, F(Ld.g1W), 256, 0, 0, 0, 1,
         h3, 256, 0, 0, 1, F(Ld.g1b), nullptr, 1.f, 0, 1, 1);             // relu
    gemm(n, 256, n, Lg, 196, 0, 0, 1, h3, 256, 0, (long long)n * 256, 0, 1,
         b1, 256, (long long)n * 256, 0, 1, nullptr, nullptr, 1.f, 0, 0, B);
    gemm(rows, 128, 256, b1, 256, 0, 0, 1, F(Ld.g2W), 128, 0, 0, 0, 1,
         curr, 128, 0, 0, 1, F(Ld.g2b), nullptr, 1.f, 1, 0, 1);

    // ================= 2x residual ChebGC =================
    auto res_cheb = [&](const float* Wc1, const float* bc1,
                        const float* Wc2, const float* bc2) {
      gemm(n, 128, n, Lc[l], LDP[l], 0, 0, 1, curr, 128, 0, s128, 0, 1,
           b1, 128, s128, 0, 1, nullptr, nullptr, 1.f, 0, 0, B);          // t = L@curr
      gemm(rows, 128, 128, curr, 128, 0, 0, 1, Wc1, 128, 0, 0, 0, 1,
           b2, 128, 0, 0, 1, bc1, nullptr, 1.f, 0, 0, 1);                 // u = curr@W0+b
      gemm(rows, 128, 128, b1, 128, 0, 0, 1, Wc1 + 128 * 128, 128, 0, 0, 0, 1,
           b2, 128, 0, 0, 1, nullptr, nullptr, 1.f, 1, 1, 1);             // u += t@W1, relu
      gemm(n, 128, n, Lc[l], LDP[l], 0, 0, 1, b2, 128, 0, s128, 0, 1,
           b1, 128, s128, 0, 1, nullptr, nullptr, 1.f, 0, 0, B);          // t = L@u
      gemm(rows, 128, 128, b2, 128, 0, 0, 1, Wc2, 128, 0, 0, 0, 1,
           h3, 128, 0, 0, 1, bc2, nullptr, 1.f, 0, 0, 1);                 // v = u@W0+b
      gemm(rows, 128, 128, b1, 128, 0, 0, 1, Wc2 + 128 * 128, 128, 0, 0, 0, 1,
           h3, 128, 0, 0, 1, nullptr, nullptr, 1.f, 1, 1, 1);             // v += t@W1, relu
      long long cnt = (long long)rows * 128;
      add_inplace<<<(unsigned)((cnt + 255) / 256), 256, 0, stream>>>(curr, h3, cnt);
    };
    res_cheb(F(Ld.r1c1W), F(Ld.r1c1b), F(Ld.r1c2W), F(Ld.r1c2b));
    res_cheb(F(Ld.r2c1W), F(Ld.r2c1b), F(Ld.r2c2W), F(Ld.r2c2b));

    // ====== GraphUnpool: transpose weight to padded layout, then fast GEMM ======
    {
      int ldpT = (n + 3) & ~3;
      long long tot = (long long)n * nnx;
      transpose_pad<<<(unsigned)((tot + 255) / 256), 256, 0, stream>>>(F(Ld.upW), upWT, n, nnx, ldpT);
      gemm(nnx, 128, n,
           upWT, ldpT, 0, 0, 1,
           curr, 128, 0, s128, 0, 1,
           spare, 128, (long long)nnx * 128, 0, 1,
           nullptr, F(Ld.upb), 1.f, 0, 0, B);
      float* tmp = curr; curr = spare; spare = tmp;
    }
  }

  // --- output ChebConv: (B,778,128) -> (B,778,3) ---
  {
    const int n = 778;
    const long long s128 = (long long)n * 128;
    float* out = (float*)d_out;
    gemm(n, 128, n, Lc[3], LDP[3], 0, 0, 1, curr, 128, 0, s128, 0, 1,
         b1, 128, s128, 0, 1, nullptr, nullptr, 1.f, 0, 0, B);            // t = L3@curr
    gemm(B * n, 3, 128, curr, 128, 0, 0, 1, F(goutW), 3, 0, 0, 0, 1,
         out, 3, 0, 0, 1, F(goutB), nullptr, 1.f, 0, 0, 1);               // y = curr@W0+b
    gemm(B * n, 3, 128, b1, 128, 0, 0, 1, F(goutW) + 128 * 3, 3, 0, 0, 0, 1,
         out, 3, 0, 0, 1, nullptr, nullptr, 1.f, 1, 0, 1);                // y += t@W1
  }
}
